// GroupedQueryAttention_64493228917130
// MI455X (gfx1250) — matrix operand order repaired and checked by reading, not compiled, not run on hardware
//
#include <hip/hip_runtime.h>

// ---------------------------------------------------------------------------
// CDNA5 (gfx1250) GQA attention: f16 WMMA GEMMs + flash attention + TDM.
// wave32; V_WMMA_F32_16X16X32_F16 throughout; TENSOR_LOAD_TO_LDS for K and V.
// ---------------------------------------------------------------------------

typedef __attribute__((ext_vector_type(16))) _Float16 v16h;
typedef __attribute__((ext_vector_type(8)))  float    v8f;
typedef __attribute__((ext_vector_type(4)))  unsigned uint32x4;
typedef __attribute__((ext_vector_type(8)))  int      int32x8;
typedef __attribute__((ext_vector_type(4)))  int      int32x4;

union AFrag { v16h h; unsigned u[8]; };
union H4    { uint2 v; _Float16 h[4]; };

__device__ __forceinline__ v8f wmma_f16(v16h a, v16h b, v8f c) {
  // 8 args: (neg_a, A, neg_b, B, c_mod, C, reuse_a, reuse_b)
  return __builtin_amdgcn_wmma_f32_16x16x32_f16(false, a, false, b, (short)0, c,
                                                false, false);
}

// A-fragment K index for 16-bit 16x32 A matrix (contiguous order):
//   v in 0..7 : K = h*16 + 2v
__device__ __forceinline__ int a_kidx(int v, int h) {
  return h * 16 + 2 * v;
}

#if __has_builtin(__builtin_amdgcn_tensor_load_to_lds)
#define HAVE_TDM 1
// DMA a 32-row x 64-half tile (row stride 512 halves) into LDS with the
// Tensor Data Mover, hardware-padding each 64-half row by 8 halves so the
// LDS image has a bank-conflict-free stride of 72 halves. One instruction.
__device__ __forceinline__ void tdm_load_tile_32x64(const _Float16* gsrc,
                                                    unsigned lds_byte) {
  const unsigned long long ga = (unsigned long long)(size_t)gsrc;
  uint32x4 g0;
  g0[0] = 1u;                                   // count=1, no gather
  g0[1] = lds_byte;                             // LDS byte address
  g0[2] = (unsigned)(ga & 0xFFFFFFFFu);         // global_addr[31:0]
  g0[3] = (unsigned)((ga >> 32) & 0x01FFFFFFu)  // global_addr[56:32]
          | (2u << 30);                         // type=2 ("image")
  int32x8 g1;
  g1[0] = (1 << 16)      // data_size = 2 bytes
        | (1 << 20)      // pad_enable
        | (4 << 22)      // pad_interval: 32 DWORDs (one 64-half row)
        | (3 << 25);     // pad_amount: 4 DWORDs (8 halves) -> stride 72
  g1[1] = (64 << 16);    // tensor_dim0 = 64 elements
  g1[2] = (32 << 16);    // tensor_dim1 = 32 rows
  g1[3] = (64 << 16);    // tile_dim0 = 64
  g1[4] = 32;            // tile_dim1 = 32 (tile_dim2 = 0)
  g1[5] = 512;           // tensor_dim0_stride = 512 elements
  g1[6] = 0; g1[7] = 0;
  const int32x4 z4 = (int32x4)0;
#if defined(__clang_major__) && (__clang_major__ >= 23)
  const int32x8 z8 = (int32x8)0;
  __builtin_amdgcn_tensor_load_to_lds(g0, g1, z4, z4, z8, 0);
#else
  __builtin_amdgcn_tensor_load_to_lds(g0, g1, z4, z4, 0);
#endif
}
#else
#define HAVE_TDM 0
#endif

// ---------------------------------------------------------------------------
// Generic C = A[M x K] @ W[K x N] + bias, f16 WMMA core.
// Block: 256 threads (8 waves), tile 128(M) x 128(N), k-chunk 64.
// Each wave: 2x4 grid of 16x16 C tiles -> 16 WMMA per k-chunk.
// ---------------------------------------------------------------------------
template <typename TA, typename TC>
__global__ __launch_bounds__(256)
void gemm_bias_wmma(const TA* __restrict__ A, const float* __restrict__ W,
                    const float* __restrict__ bias, TC* __restrict__ C,
                    int N, int K) {
  __shared__ _Float16 lds_a[128 * 72];    // [m][k]  stride 72 (64 + 8 pad)
  __shared__ _Float16 lds_b[64 * 136];    // [k][n]  stride 136 (128 + 8 pad)

  const int tid  = threadIdx.x;
  const int lane = tid & 31;
  const int wave = tid >> 5;       // 0..7
  const int wm   = wave >> 1;      // 0..3: 32-row group
  const int wn   = wave & 1;       // 0..1: 64-col group
  const int hh   = lane >> 4;      // lane half
  const int ln   = lane & 15;

  const int m0 = blockIdx.y * 128;
  const int n0 = blockIdx.x * 128;

  v8f acc[2][4];
#pragma unroll
  for (int i = 0; i < 2; ++i)
#pragma unroll
    for (int j = 0; j < 4; ++j) acc[i][j] = (v8f)(0.0f);

  for (int kc = 0; kc < K; kc += 64) {
    // ---- stage A tile 128x64 (row-major, convert to f16, uint2 stores) ----
    {
      const int row0 = tid >> 4;           // 0..15
      const int col  = (tid & 15) * 4;     // 0..60
#pragma unroll
      for (int p = 0; p < 8; ++p) {
        const int r = row0 + p * 16;
        const TA* src = A + (size_t)(m0 + r) * K + kc + col;
        if constexpr (sizeof(TA) == 4) {
          const float4 f = *(const float4*)src;
          H4 t;
          t.h[0] = (_Float16)f.x; t.h[1] = (_Float16)f.y;
          t.h[2] = (_Float16)f.z; t.h[3] = (_Float16)f.w;
          *(uint2*)&lds_a[r * 72 + col] = t.v;
        } else {
          *(uint2*)&lds_a[r * 72 + col] = *(const uint2*)src;
        }
      }
    }
    // ---- stage W tile 64x128 (row-major [k][n], uint2 stores) ----
    {
      const int k  = tid >> 2;             // 0..63
      const int nb = (tid & 3) * 32;       // 0,32,64,96
      const float* src = W + (size_t)(kc + k) * N + n0 + nb;
#pragma unroll
      for (int j = 0; j < 32; j += 4) {
        const float4 f = *(const float4*)(src + j);
        H4 t;
        t.h[0] = (_Float16)f.x; t.h[1] = (_Float16)f.y;
        t.h[2] = (_Float16)f.z; t.h[3] = (_Float16)f.w;
        *(uint2*)&lds_b[k * 136 + nb + j] = t.v;
      }
    }
    __syncthreads();

    if (kc + 64 < K) {  // speculative prefetch of next tiles (global_prefetch_b8)
      __builtin_prefetch(A + (size_t)(m0 + (tid >> 1)) * K + kc + 64, 0, 1);
      __builtin_prefetch(W + (size_t)(kc + 64 + (tid >> 2)) * N + n0, 0, 1);
    }

#pragma unroll
    for (int ch = 0; ch < 2; ++ch) {       // two 32-wide K sub-chunks
      AFrag af[2], bf[4];
#pragma unroll
      for (int rt = 0; rt < 2; ++rt) {
        const int row = wm * 32 + rt * 16 + ln;
#pragma unroll
        for (int v = 0; v < 8; ++v)
          af[rt].u[v] =
              *(const unsigned*)&lds_a[row * 72 + ch * 32 + a_kidx(v, hh)];
      }
#pragma unroll
      for (int nt = 0; nt < 4; ++nt) {
        const int n = wn * 64 + nt * 16 + ln;
#pragma unroll
        for (int v = 0; v < 8; ++v) {
          const int kk = ch * 32 + hh * 16 + 2 * v;
          const unsigned lo = *(const unsigned short*)&lds_b[kk * 136 + n];
          const unsigned hi = *(const unsigned short*)&lds_b[(kk + 1) * 136 + n];
          bf[nt].u[v] = lo | (hi << 16);
        }
      }
#pragma unroll
      for (int rt = 0; rt < 2; ++rt)
#pragma unroll
        for (int nt = 0; nt < 4; ++nt)
          acc[rt][nt] = wmma_f16(af[rt].h, bf[nt].h, acc[rt][nt]);
    }
    __syncthreads();
  }

  // ---- epilogue: bias + store (C layout: M = r + 8h, N = ln) ----
#pragma unroll
  for (int rt = 0; rt < 2; ++rt)
#pragma unroll
    for (int nt = 0; nt < 4; ++nt) {
      const int n  = n0 + wn * 64 + nt * 16 + ln;
      const float bb = bias[n];
#pragma unroll
      for (int r = 0; r < 8; ++r) {
        const int m = m0 + wm * 32 + rt * 16 + r + 8 * hh;
        C[(size_t)m * N + n] = (TC)(acc[rt][nt][r] + bb);
      }
    }
}

// ---------------------------------------------------------------------------
// RoPE (rotate-half), positions 1..t, in-place on f16 (b,t,h,64) tensor.
// ---------------------------------------------------------------------------
__global__ void rope_kernel(_Float16* __restrict__ x, int nheads, int total) {
  const int idx = blockIdx.x * blockDim.x + threadIdx.x;
  if (idx >= total) return;
  const int i    = idx & 31;       // pair index 0..31
  const int rest = idx >> 5;       // (b*1024 + t)*nheads + h
  const int t    = (rest / nheads) & 1023;
  _Float16* p = x + (size_t)rest * 64;
  const float theta = __powf(10000.0f, -(float)(2 * i) / 64.0f);
  const float ang = (float)(t + 1) * theta;
  const float c = __cosf(ang), s = __sinf(ang);
  const float x0 = (float)p[i], x1 = (float)p[i + 32];
  p[i]      = (_Float16)(x0 * c - x1 * s);
  p[i + 32] = (_Float16)(x1 * c + x0 * s);
}

// ---------------------------------------------------------------------------
// Flash attention, causal, GQA (32 q heads / 8 kv heads), HD=64.
// Block: 128 threads (4 waves) owns 128 q rows of one (b, qhead).
// K and V tiles both staged row-major by the Tensor Data Mover (LDS-padded
// to stride 72); mask applied only on the diagonal key block.
// ---------------------------------------------------------------------------
__global__ __launch_bounds__(128)
void flash_attn_wmma(const _Float16* __restrict__ q,
                     const _Float16* __restrict__ k,
                     const _Float16* __restrict__ v,
                     _Float16* __restrict__ y) {
  constexpr int T = 1024, EQ = 2048, EKV = 512;
  const int qb = blockIdx.x;          // 0..7 (128 q rows each)
  const int hq = blockIdx.y;          // 0..31
  const int b  = blockIdx.z;          // 0..1
  const int hk = hq >> 2;             // GQA: 4 q heads per kv head

  const _Float16* qp = q + (size_t)b * T * EQ  + hq * 64;
  const _Float16* kp = k + (size_t)b * T * EKV + hk * 64;
  const _Float16* vp = v + (size_t)b * T * EKV + hk * 64;
  _Float16*       yp = y + (size_t)b * T * EQ  + hq * 64;

  __shared__ _Float16 lds_k[32 * 72];        // [key][d], stride 72
  __shared__ _Float16 lds_v[32 * 72];        // [key][d], stride 72 (row-major)
  __shared__ _Float16 lds_p[4][16 * 40];     // per-wave P bounce [m][key]

  const int tid  = threadIdx.x;
  const int lane = tid & 31;
  const int wave = tid >> 5;       // 0..3
  const int hh   = lane >> 4;
  const int ln   = lane & 15;
  const int qw0  = qb * 128 + wave * 32;     // first q row of this wave

  // ---- Q fragments straight from global (A layout, rows contiguous in d) ----
  AFrag qa[2][2];
#pragma unroll
  for (int rt = 0; rt < 2; ++rt) {
    const _Float16* row = qp + (size_t)(qw0 + rt * 16 + ln) * EQ;
#pragma unroll
    for (int ch = 0; ch < 2; ++ch)
#pragma unroll
      for (int v = 0; v < 8; ++v)
        qa[rt][ch].u[v] = *(const unsigned*)(row + ch * 32 + a_kidx(v, hh));
  }

  v8f acc[2][4];
  float mrow[2][8], lrow[2][8];
#pragma unroll
  for (int rt = 0; rt < 2; ++rt) {
#pragma unroll
    for (int dt = 0; dt < 4; ++dt) acc[rt][dt] = (v8f)(0.0f);
#pragma unroll
    for (int r = 0; r < 8; ++r) { mrow[rt][r] = -1e30f; lrow[rt][r] = 0.0f; }
  }

  const int kb_max_blk  = qb * 4 + 3;     // block causal bound (inclusive)
  const int kb_max_wave = qw0 >> 5;       // this wave's causal bound

  for (int kb = 0; kb <= kb_max_blk; ++kb) {
    // ---- stage K and V tiles (rows kb*32..kb*32+31) ----
#if HAVE_TDM
    if (wave == 0) {   // one wave issues both DMAs; they retire in order
      tdm_load_tile_32x64(kp + (size_t)(kb * 32) * EKV,
                          (unsigned)(size_t)&lds_k[0]);
      tdm_load_tile_32x64(vp + (size_t)(kb * 32) * EKV,
                          (unsigned)(size_t)&lds_v[0]);
      __builtin_amdgcn_s_wait_tensorcnt(0);
    }
#else
    {  // fallback: vectorized manual staging (row-major, no transpose needed)
      const int key = tid >> 2;            // 0..31
      const int d0  = (tid & 3) * 16;      // 0,16,32,48
      const _Float16* ks = kp + (size_t)(kb * 32 + key) * EKV + d0;
      *(uint4*)&lds_k[key * 72 + d0]     = ((const uint4*)ks)[0];
      *(uint4*)&lds_k[key * 72 + d0 + 8] = ((const uint4*)ks)[1];
      const _Float16* vs = vp + (size_t)(kb * 32 + key) * EKV + d0;
      *(uint4*)&lds_v[key * 72 + d0]     = ((const uint4*)vs)[0];
      *(uint4*)&lds_v[key * 72 + d0 + 8] = ((const uint4*)vs)[1];
    }
#endif
    __syncthreads();

    if (kb <= kb_max_wave) {
      // K^T B-fragments: element (K=d, N=key); d contiguous in lds_k rows.
      AFrag bk[2][2];
#pragma unroll
      for (int ct = 0; ct < 2; ++ct)
#pragma unroll
        for (int ch = 0; ch < 2; ++ch)
#pragma unroll
          for (int v = 0; v < 8; ++v)
            bk[ct][ch].u[v] = *(const unsigned*)
                &lds_k[(ct * 16 + ln) * 72 + ch * 32 + hh * 16 + 2 * v];
      // V B-fragments: element (K=key, N=d); V is row-major, so pack two
      // u16 reads (rows kk, kk+1 at column d) per operand dword.
      AFrag bv[4];
#pragma unroll
      for (int dt = 0; dt < 4; ++dt)
#pragma unroll
        for (int v = 0; v < 8; ++v) {
          const int kk = hh * 16 + 2 * v;
          const unsigned lo =
              *(const unsigned short*)&lds_v[kk * 72 + dt * 16 + ln];
          const unsigned hi =
              *(const unsigned short*)&lds_v[(kk + 1) * 72 + dt * 16 + ln];
          bv[dt].u[v] = lo | (hi << 16);
        }

      const bool diag = (kb == kb_max_wave);   // only diagonal block masks

#pragma unroll
      for (int rt = 0; rt < 2; ++rt) {
        // S = (Q K^T) / 8 over this 16x32 tile (two 16x16 column tiles)
        v8f s[2];
#pragma unroll
        for (int ct = 0; ct < 2; ++ct) {
          s[ct] = (v8f)(0.0f);
#pragma unroll
          for (int ch = 0; ch < 2; ++ch)
            s[ct] = wmma_f16(qa[rt][ch].h, bk[ct][ch].h, s[ct]);
#pragma unroll
          for (int r = 0; r < 8; ++r) s[ct][r] *= 0.125f;
        }
        if (diag) {
          const int qbase = qw0 + rt * 16 + 8 * hh;   // + r gives q index
#pragma unroll
          for (int ct = 0; ct < 2; ++ct) {
            const int kj = kb * 32 + ct * 16 + ln;
#pragma unroll
            for (int r = 0; r < 8; ++r)
              if (kj > qbase + r) s[ct][r] = -1e30f;  // causal mask
          }
        }
        // online softmax; row of S spans the 16 lanes of a half-wave
#pragma unroll
        for (int r = 0; r < 8; ++r) {
          float mx = fmaxf(s[0][r], s[1][r]);
          mx = fmaxf(mx, __shfl_xor(mx, 1, 32));
          mx = fmaxf(mx, __shfl_xor(mx, 2, 32));
          mx = fmaxf(mx, __shfl_xor(mx, 4, 32));
          mx = fmaxf(mx, __shfl_xor(mx, 8, 32));
          const float mnew  = fmaxf(mrow[rt][r], mx);
          const float alpha = __expf(mrow[rt][r] - mnew);
          mrow[rt][r] = mnew;
          const float p0 = __expf(s[0][r] - mnew);
          const float p1 = __expf(s[1][r] - mnew);
          float ps = p0 + p1;
          ps += __shfl_xor(ps, 1, 32);
          ps += __shfl_xor(ps, 2, 32);
          ps += __shfl_xor(ps, 4, 32);
          ps += __shfl_xor(ps, 8, 32);
          lrow[rt][r] = lrow[rt][r] * alpha + ps;
#pragma unroll
          for (int dt = 0; dt < 4; ++dt) acc[rt][dt][r] *= alpha;
          s[0][r] = p0; s[1][r] = p1;
        }
        // bounce P through LDS: C layout -> A layout (wave-private region)
#pragma unroll
        for (int ct = 0; ct < 2; ++ct)
#pragma unroll
          for (int r = 0; r < 8; ++r)
            lds_p[wave][(r + 8 * hh) * 40 + ct * 16 + ln] = (_Float16)s[ct][r];
        AFrag pa;
#pragma unroll
        for (int v = 0; v < 8; ++v)
          pa.u[v] = *(const unsigned*)&lds_p[wave][ln * 40 + a_kidx(v, hh)];
        // O += P @ V
#pragma unroll
        for (int dt = 0; dt < 4; ++dt)
          acc[rt][dt] = wmma_f16(pa.h, bv[dt].h, acc[rt][dt]);
      }
    }
    __syncthreads();
  }

  // ---- normalize and write y (b,t,h,d) in f16 ----
#pragma unroll
  for (int rt = 0; rt < 2; ++rt)
#pragma unroll
    for (int dt = 0; dt < 4; ++dt)
#pragma unroll
      for (int r = 0; r < 8; ++r) {
        const int tq = qw0 + rt * 16 + r + 8 * hh;
        const float o = acc[rt][dt][r] / lrow[rt][r];
        yp[(size_t)tq * EQ + dt * 16 + ln] = (_Float16)o;
      }
}

// ---------------------------------------------------------------------------
// Launcher
// ---------------------------------------------------------------------------
extern "C" void kernel_launch(void* const* d_in, const int* in_sizes, int n_in,
                              void* d_out, int out_size, void* d_ws, size_t ws_size,
                              hipStream_t stream) {
  const float* x  = (const float*)d_in[0];
  const float* wq = (const float*)d_in[1];
  const float* bq = (const float*)d_in[2];
  const float* wk = (const float*)d_in[3];
  const float* bk = (const float*)d_in[4];
  const float* wv = (const float*)d_in[5];
  const float* bv = (const float*)d_in[6];
  const float* wo = (const float*)d_in[7];
  const float* bo = (const float*)d_in[8];
  float* out = (float*)d_out;

  // workspace: f16 q (8MB) | k (2MB) | v (2MB) | y (8MB)
  _Float16* qbuf = (_Float16*)d_ws;
  _Float16* kbuf = qbuf + (size_t)2 * 1024 * 2048;
  _Float16* vbuf = kbuf + (size_t)2 * 1024 * 512;
  _Float16* ybuf = vbuf + (size_t)2 * 1024 * 512;

  const dim3 blk(256);
  // QKV projections (M = B*T = 2048, K = 2048)
  gemm_bias_wmma<float, _Float16><<<dim3(16, 16), blk, 0, stream>>>(x, wq, bq, qbuf, 2048, 2048);
  gemm_bias_wmma<float, _Float16><<<dim3(4, 16),  blk, 0, stream>>>(x, wk, bk, kbuf, 512, 2048);
  gemm_bias_wmma<float, _Float16><<<dim3(4, 16),  blk, 0, stream>>>(x, wv, bv, vbuf, 512, 2048);
  // RoPE on q (32 heads) and k (8 heads)
  rope_kernel<<<(2097152 + 255) / 256, 256, 0, stream>>>(qbuf, 32, 2097152);
  rope_kernel<<<(524288 + 255) / 256, 256, 0, stream>>>(kbuf, 8, 524288);
  // flash attention: 8 q-blocks x 32 heads x 2 batches
  flash_attn_wmma<<<dim3(8, 32, 2), dim3(128), 0, stream>>>(qbuf, kbuf, vbuf, ybuf);
  // output projection -> fp32 out
  gemm_bias_wmma<_Float16, float><<<dim3(16, 16), blk, 0, stream>>>(ybuf, wo, bo, out, 2048, 2048);
}
